// MixtureOfExperts_78477642432589
// MI455X (gfx1250) — matrix operand order repaired and checked by reading, not compiled, not run on hardware
//
#include <hip/hip_runtime.h>

// ---------------------------------------------------------------------------
// MoE top-2 forward for MI455X (gfx1250, wave32, WMMA bf16).
// T=8192 tokens, D=512, H=1024, E=8 experts, K=2.
// Sparse routing (only top-2 experts computed per token):
//   - v_wmma_f32_16x16x32_bf16 for both expert GEMMs, fp32 accumulate
//   - global_load_async_to_lds_b128 (ASYNCcnt) for the token-row gather
// ---------------------------------------------------------------------------

#define T_TOK 8192
#define D_IN  512
#define H_HID 1024
#define E_EXP 8
#define MT    32          // tokens per workgroup tile
#define XSTR  520         // LDS row stride (bf16) for X tile: 512 + 8 pad
#define HSTR  136         // LDS row stride (bf16) for H chunk: 128 + 8 pad

typedef __attribute__((ext_vector_type(16))) __bf16 v16bf;
typedef __attribute__((ext_vector_type(8)))  __bf16 v8bf;
typedef __attribute__((ext_vector_type(8)))  float  v8f;

union V16U { v16bf v; v8bf h[2]; };

__device__ __forceinline__ v8f wmma_bf16(v16bf a, v16bf b, v8f c) {
  // 8 args: (neg_a, A, neg_b, B, c_mod, C, reuse_a, reuse_b)
  return __builtin_amdgcn_wmma_f32_16x16x32_bf16(false, a, false, b,
                                                 (short)0, c, false, false);
}

// Async 16B copy global -> LDS (per-lane, tracked by ASYNCcnt).
// Generic LDS pointer's low 32 bits are the LDS byte offset (aperture rule).
__device__ __forceinline__ void async_ld_b128(const void* gptr, void* lptr) {
  const unsigned int       l = (unsigned int)(unsigned long long)(uintptr_t)lptr;
  const unsigned long long g = (unsigned long long)(uintptr_t)gptr;
  asm volatile("global_load_async_to_lds_b128 %0, %1, off"
               :: "v"(l), "v"(g) : "memory");
}
__device__ __forceinline__ void wait_async0() {
  asm volatile("s_wait_asynccnt 0" ::: "memory");
}

// A fragment (16x32 bf16, M x K) from a row-major LDS tile.
// ISA layout: lanes 0-15 = row M=lane, K in {k..k+7, k+16..k+23};
//             lanes 16-31 = row M=lane-16, K in {k+8..k+15, k+24..k+31}.
__device__ __forceinline__ v16bf load_a_lds(const __bf16* row, int k, int lane) {
  const int kb = k + ((lane & 16) ? 8 : 0);
  V16U u;
  u.h[0] = *(const v8bf*)(row + kb);        // 16B ds_load
  u.h[1] = *(const v8bf*)(row + kb + 16);   // 16B ds_load
  return u.v;
}

// B fragment (32x16 bf16, K x N) from a transposed (N-major) global matrix:
// `colrow` points at the start of column n's K-run. Lanes 0-15 (col = lane)
// hold K in {k..k+7, k+16..k+23}, lanes 16-31 hold K in {k+8..k+15, k+24..k+31}.
__device__ __forceinline__ v16bf load_b_gl(const __bf16* colrow, int k, int lane) {
  const int kb = k + ((lane & 16) ? 8 : 0);
  V16U u;
  u.h[0] = *(const v8bf*)(colrow + kb);     // global_load_b128
  u.h[1] = *(const v8bf*)(colrow + kb + 16);
  return u.v;
}

// ---------------------------------------------------------------------------
__global__ __launch_bounds__(256) void init_kernel(int* cnt, float* imp) {
  const int i = threadIdx.x;
  if (i < E_EXP) { cnt[i] = 0; imp[i] = 0.0f; }
}

__global__ __launch_bounds__(256) void zero_out_kernel(float4* __restrict__ out4) {
  out4[(size_t)blockIdx.x * 256 + threadIdx.x] = make_float4(0.f, 0.f, 0.f, 0.f);
}

// x [T][D] fp32 -> Xbf [T][D] bf16
__global__ __launch_bounds__(256) void cvt_x_kernel(const float* __restrict__ x,
                                                    __bf16* __restrict__ Xbf) {
  const size_t i = (size_t)blockIdx.x * 256 + threadIdx.x;
  Xbf[i] = (__bf16)x[i];
}

// W1 [E][D][H] fp32 -> W1T [E][H][D] bf16   (i = e*2^19 + h*512 + d)
__global__ __launch_bounds__(256) void cvt_w1_kernel(const float* __restrict__ W1,
                                                     __bf16* __restrict__ W1T) {
  const size_t i = (size_t)blockIdx.x * 256 + threadIdx.x;
  const int e   = (int)(i >> 19);
  const int rem = (int)(i & ((1u << 19) - 1));
  const int h   = rem >> 9;
  const int d   = rem & 511;
  W1T[i] = (__bf16)W1[((size_t)e << 19) + (size_t)d * H_HID + h];
}

// W2 [E][H][D] fp32 -> W2T [E][D][H] bf16   (i = e*2^19 + d*1024 + h)
__global__ __launch_bounds__(256) void cvt_w2_kernel(const float* __restrict__ W2,
                                                     __bf16* __restrict__ W2T) {
  const size_t i = (size_t)blockIdx.x * 256 + threadIdx.x;
  const int e   = (int)(i >> 19);
  const int rem = (int)(i & ((1u << 19) - 1));
  const int d   = rem >> 10;
  const int h   = rem & 1023;
  W2T[i] = (__bf16)W2[((size_t)e << 19) + (size_t)h * D_IN + d];
}

// ---------------------------------------------------------------------------
// Gating: one wave (32 lanes) per token. logits = x[t] @ Wg, top-2, softmax,
// routing lists + importance/load statistics.
__global__ __launch_bounds__(256) void gating_kernel(
    const float* __restrict__ x, const float* __restrict__ Wg,
    int* __restrict__ tok_list, float* __restrict__ prob_list,
    int* __restrict__ cnt, float* __restrict__ importance) {
  const int t    = blockIdx.x * 8 + (threadIdx.x >> 5);
  const int lane = threadIdx.x & 31;
  const float* xr = x + (size_t)t * D_IN;

  float acc[E_EXP];
  #pragma unroll
  for (int e = 0; e < E_EXP; ++e) acc[e] = 0.0f;

  for (int d = lane; d < D_IN; d += 32) {
    const float xv = xr[d];
    const float* wr = Wg + (size_t)d * E_EXP;   // Wg is [D][E] row-major
    #pragma unroll
    for (int e = 0; e < E_EXP; ++e) acc[e] += xv * wr[e];
  }
  #pragma unroll
  for (int e = 0; e < E_EXP; ++e)
    for (int off = 16; off > 0; off >>= 1)
      acc[e] += __shfl_xor(acc[e], off, 32);

  if (lane == 0) {
    int e0 = 0; float v0 = acc[0];
    #pragma unroll
    for (int e = 1; e < E_EXP; ++e)
      if (acc[e] > v0) { v0 = acc[e]; e0 = e; }     // strict > keeps first (jax tie rule)
    int e1 = -1; float v1 = -3.4e38f;
    #pragma unroll
    for (int e = 0; e < E_EXP; ++e)
      if (e != e0 && acc[e] > v1) { v1 = acc[e]; e1 = e; }
    const float p0 = 1.0f / (1.0f + __expf(v1 - v0));  // softmax over [v0,v1]
    const float p1 = 1.0f - p0;

    const int pos0 = atomicAdd(&cnt[e0], 1);
    tok_list[e0 * T_TOK + pos0]  = t;
    prob_list[e0 * T_TOK + pos0] = p0;
    const int pos1 = atomicAdd(&cnt[e1], 1);
    tok_list[e1 * T_TOK + pos1]  = t;
    prob_list[e1 * T_TOK + pos1] = p1;
    atomicAdd(&importance[e0], p0);
    atomicAdd(&importance[e1], p1);
  }
}

// Losses: var(ddof=1)/(mean^2 + eps) for importance and per-expert load.
__global__ void loss_kernel(const float* __restrict__ imp,
                            const int* __restrict__ cnt,
                            float* __restrict__ outs) {
  if (threadIdx.x == 0) {
    float s = 0.0f;
    #pragma unroll
    for (int e = 0; e < E_EXP; ++e) s += imp[e];
    float mean = s / (float)E_EXP, var = 0.0f;
    #pragma unroll
    for (int e = 0; e < E_EXP; ++e) { float d = imp[e] - mean; var += d * d; }
    var /= (float)(E_EXP - 1);
    outs[0] = var / (mean * mean + 1e-10f);

    s = 0.0f;
    #pragma unroll
    for (int e = 0; e < E_EXP; ++e) s += (float)cnt[e];
    mean = s / (float)E_EXP; var = 0.0f;
    #pragma unroll
    for (int e = 0; e < E_EXP; ++e) { float d = (float)cnt[e] - mean; var += d * d; }
    var /= (float)(E_EXP - 1);
    outs[1] = var / (mean * mean + 1e-10f);
  }
}

// ---------------------------------------------------------------------------
// Fused expert kernel: one workgroup = (expert e, 32-token tile).
// 8 waves. X rows gathered with async global->LDS copies; GEMM1
// relu(X @ W1 + b1) streamed in 128-wide H chunks through LDS; GEMM2
// accumulates O(32x512) in registers; epilogue scatter-adds p*(O+b2).
__global__ __launch_bounds__(256) void moe_expert_kernel(
    const __bf16* __restrict__ Xbf, const float* __restrict__ b1,
    const float* __restrict__ b2, const __bf16* __restrict__ W1T,
    const __bf16* __restrict__ W2T, const int* __restrict__ tok_list,
    const float* __restrict__ prob_list, const int* __restrict__ cnt,
    float* __restrict__ out) {
  __shared__ __bf16 Xs[MT * XSTR];   // 33,280 B
  __shared__ __bf16 Hs[MT * HSTR];   //  8,704 B
  __shared__ int   toks[MT];
  __shared__ float probs[MT];

  const int e    = blockIdx.x & (E_EXP - 1);
  const int tile = blockIdx.x >> 3;
  const int n    = cnt[e];
  const int m0   = tile * MT;
  if (m0 >= n) return;               // uniform per block: EXEC stays all-ones

  const int tid  = threadIdx.x;
  const int wave = tid >> 5;
  const int lane = tid & 31;
  const int lrow = lane & 15;
  const int mh   = wave & 1;         // which 16-row half of the 32-token tile
  const int nq   = wave >> 1;        // 0..3: N quadrant
  const int rb   = (lane & 16) ? 8 : 0;  // C/D: lanes 16-31 hold M = r+8

  if (tid < MT) {
    const int r = m0 + tid;
    if (r < n) { toks[tid] = tok_list[e * T_TOK + r]; probs[tid] = prob_list[e * T_TOK + r]; }
    else       { toks[tid] = 0;                       probs[tid] = 0.0f; }  // p=0 dummy
  }
  __syncthreads();

  // Async gather: 32 rows x 1024B, 8 threads/row, 8 x 16B transfers each.
  {
    const int row = tid >> 3;
    const int s8  = tid & 7;
    const __bf16* g = Xbf + (size_t)toks[row] * D_IN;
    __bf16* l = &Xs[row * XSTR];
    #pragma unroll
    for (int j = 0; j < 8; ++j) {
      const int c = (s8 + 8 * j) * 8;   // bf16 elements; 16B aligned
      async_ld_b128(g + c, l + c);
    }
    wait_async0();
  }
  __syncthreads();

  v8f acc2[8] = {};                  // O tile: wave owns 16(M) x 128(N)

  const __bf16* arow1 = &Xs[(mh * 16 + lrow) * XSTR];
  const __bf16* arow2 = &Hs[(mh * 16 + lrow) * HSTR];

  for (int hc = 0; hc < H_HID / 128; ++hc) {
    // ---- GEMM1: Hc[32x128] = relu(X[32x512] @ W1[:, hc*128 : +128] + b1)
    const int ncol0 = hc * 128 + nq * 32 + lrow;
    const __bf16* bp0 = W1T + ((size_t)(e * H_HID + ncol0)) * D_IN;
    const __bf16* bp1 = bp0 + (size_t)16 * D_IN;
    v8f h0 = {}, h1 = {};
    #pragma unroll 4
    for (int k = 0; k < D_IN; k += 32) {
      const v16bf a = load_a_lds(arow1, k, lane);
      h0 = wmma_bf16(a, load_b_gl(bp0, k, lane), h0);
      h1 = wmma_bf16(a, load_b_gl(bp1, k, lane), h1);
    }
    const float bs0 = b1[e * H_HID + ncol0];
    const float bs1 = b1[e * H_HID + ncol0 + 16];
    #pragma unroll
    for (int r = 0; r < 8; ++r) {
      Hs[(mh * 16 + rb + r) * HSTR + nq * 32 + lrow]      = (__bf16)fmaxf(h0[r] + bs0, 0.0f);
      Hs[(mh * 16 + rb + r) * HSTR + nq * 32 + 16 + lrow] = (__bf16)fmaxf(h1[r] + bs1, 0.0f);
    }
    __syncthreads();

    // ---- GEMM2 partial: O[32x512] += Hc[32x128] @ W2[hc*128 : +128, :]
    v16bf af[4];
    #pragma unroll
    for (int kk = 0; kk < 4; ++kk) af[kk] = load_a_lds(arow2, kk * 32, lane);
    #pragma unroll
    for (int t = 0; t < 8; ++t) {
      const __bf16* b2p =
          W2T + ((size_t)(e * D_IN + nq * 128 + t * 16 + lrow)) * H_HID + hc * 128;
      #pragma unroll
      for (int kk = 0; kk < 4; ++kk)
        acc2[t] = wmma_bf16(af[kk], load_b_gl(b2p, kk * 32, lane), acc2[t]);
    }
    __syncthreads();                 // before Hs is overwritten next chunk
  }

  // ---- Combine: out[tok] += p * (O + b2). Two experts/token -> atomics.
  #pragma unroll
  for (int t = 0; t < 8; ++t) {
    const int col = nq * 128 + t * 16 + lrow;
    const float bb = b2[e * D_IN + col];
    #pragma unroll
    for (int r = 0; r < 8; ++r) {
      const int row = mh * 16 + rb + r;
      atomicAdd(out + (size_t)toks[row] * D_IN + col, probs[row] * (acc2[t][r] + bb));
    }
  }
}

// ---------------------------------------------------------------------------
extern "C" void kernel_launch(void* const* d_in, const int* in_sizes, int n_in,
                              void* d_out, int out_size, void* d_ws, size_t ws_size,
                              hipStream_t stream) {
  (void)in_sizes; (void)n_in; (void)out_size; (void)ws_size;
  const float* x  = (const float*)d_in[0];   // [T,D]
  const float* Wg = (const float*)d_in[1];   // [D,E]
  const float* W1 = (const float*)d_in[2];   // [E,D,H]
  const float* b1 = (const float*)d_in[3];   // [E,H]
  const float* W2 = (const float*)d_in[4];   // [E,H,D]
  const float* b2 = (const float*)d_in[5];   // [E,D]
  float* out = (float*)d_out;                // [T*D] ++ [imp_loss, load_loss]

  // Workspace layout (~25.7 MB)
  char* w = (char*)d_ws;
  __bf16* W1T      = (__bf16*)(w);                       // 8,388,608 B
  __bf16* W2T      = (__bf16*)(w + 8388608);             // 8,388,608 B
  __bf16* Xbf      = (__bf16*)(w + 16777216);            // 8,388,608 B
  int*    tok_list = (int*)  (w + 25165824);             //   262,144 B
  float*  prob_list= (float*)(w + 25427968);             //   262,144 B
  int*    cnt      = (int*)  (w + 25690112);             //        32 B (pad 64)
  float*  imp      = (float*)(w + 25690176);             //        32 B

  init_kernel<<<1, 256, 0, stream>>>(cnt, imp);
  zero_out_kernel<<<(T_TOK * D_IN / 4) / 256, 256, 0, stream>>>((float4*)out);
  cvt_x_kernel<<<(T_TOK * D_IN) / 256, 256, 0, stream>>>(x, Xbf);
  cvt_w1_kernel<<<(E_EXP * D_IN * H_HID) / 256, 256, 0, stream>>>(W1, W1T);
  cvt_w2_kernel<<<(E_EXP * D_IN * H_HID) / 256, 256, 0, stream>>>(W2, W2T);
  gating_kernel<<<T_TOK / 8, 256, 0, stream>>>(x, Wg, tok_list, prob_list, cnt, imp);
  loss_kernel<<<1, 32, 0, stream>>>(imp, cnt, out + (size_t)T_TOK * D_IN);
  // worst case one expert takes all 8192 tokens -> 256 tile slots per expert
  moe_expert_kernel<<<E_EXP * (T_TOK / MT), 256, 0, stream>>>(
      Xbf, b1, b2, W1T, W2T, tok_list, prob_list, cnt, out);
}
